// QuantumDMRGLayer_29987461661225
// MI455X (gfx1250) — matrix operand order repaired and checked by reading, not compiled, not run on hardware
//
#include <hip/hip_runtime.h>
#include <hip/hip_bf16.h>
#include <stddef.h>

// ---------------------------------------------------------------------------
// MPS chain contraction for MI455X (gfx1250), wave32 + WMMA bf16 + TDM.
//   carry_new[b,:] = x0[b]*(carry[b,:] @ T0) + x1[b]*(carry[b,:] @ T1)
// Each workgroup owns a 16-row batch slab and runs the sequential chain with
// carry resident in LDS (bf16 operands, f32 WMMA accumulation). Weights are
// pre-converted/transposed to bf16 [n][k] in d_ws; the 192MB L2 holds all
// 66MB of bf16 cores, so per-step staging hits L2. Per step, ONE Tensor Data
// Mover descriptor DMA's the whole 2x256x256 bf16 core into LDS with the
// padded pitch applied in hardware (pad_interval=128 DWORDs, pad_amount=4
// DWORDs -> 264-element rows), tracked on TENSORcnt.
// Dynamic LDS = ~288KB/WG (within CDNA5's 320KB WGP budget).
// ---------------------------------------------------------------------------

typedef __bf16 bf16_t;
typedef __bf16 v16bf __attribute__((ext_vector_type(16)));
typedef __bf16 v8bf  __attribute__((ext_vector_type(8)));
typedef float  v8f   __attribute__((ext_vector_type(8)));
typedef unsigned int v4u __attribute__((ext_vector_type(4)));
typedef int          v8i __attribute__((ext_vector_type(8)));
typedef int          v4i __attribute__((ext_vector_type(4)));

#define D          256
#define NLAB       10
#define NSTEP_PRE  127
#define NSTEP_POST 126
#define NMID       253
#define TPITCH     264   // LDS T row pitch: 256 data + 8 pad bf16 (16B-aligned rows)
#define CPITCH     264   // LDS carry row pitch

// dyn-LDS: TtL[2][256][TPITCH] bf16 | carryL[2][16][CPITCH] bf16 | xls[32] f32 | red[256] f32
#define SMEM_BYTES ((2*D*TPITCH + 2*16*CPITCH) * 2 + (32 + 256) * 4)

#if defined(__AMDGCN__) && __has_builtin(__builtin_amdgcn_tensor_load_to_lds) && \
    __has_builtin(__builtin_amdgcn_s_wait_tensorcnt)
#define TDM_OK 1
#endif

// Build a v16bf fragment from two 16-byte LDS chunks.
static __device__ __forceinline__ v16bf frag2(const bf16_t* lo, const bf16_t* hi) {
  union { v16bf v; v8bf h[2]; } u;
  u.h[0] = *reinterpret_cast<const v8bf*>(lo);
  u.h[1] = *reinterpret_cast<const v8bf*>(hi);
  return u.v;
}

#ifdef TDM_OK
// One TDM descriptor: 3-D tile x=256 elems, y=256 rows, z=2 (physical index),
// data_size=2B, z-stride = pstride elements; LDS padding 4 DWORDs per 128.
static __device__ __forceinline__ void tdm_load_T(const bf16_t* src, size_t pstride,
                                                  unsigned ldsOff) {
  unsigned long long ga = (unsigned long long)(uintptr_t)src;
  v4u g0 = { 1u,                                   // count=1 (valid), user mode
             ldsOff,                               // lds_addr (bytes)
             (unsigned)(ga & 0xFFFFFFFFu),         // global_addr[31:0]
             (unsigned)((ga >> 32) & 0x01FFFFFFu) | (2u << 30) };  // addr[56:32] | type=2
  v8i g1 = { (int)((1u << 16) | (1u << 20) | (6u << 22) | (3u << 25)),
             // ^ data_size=1(2B), pad_enable, pad_interval=6(128 DW), pad_amount=3(4 DW)
             (int)(256u << 16),                    // tensor_dim0 lo16 in bits 63:48
             (int)(256u << 16),                    // tensor_dim0 hi=0 | tensor_dim1 lo16=256
             (int)(256u << 16),                    // tensor_dim1 hi=0 | tile_dim0=256
             (int)(256u | (2u << 16)),             // tile_dim1=256 | tile_dim2=2
             (int)256,                             // tensor_dim0_stride lo32 = 256
             (int)((pstride & 0xFFFFu) << 16),     // dim0_stride hi=0 | dim1_stride lo16
             (int)(pstride >> 16) };               // tensor_dim1_stride[47:16]
  v4i g2 = { 2, 0, 0, 0 };                         // tensor_dim2 = 2
  v4i g3 = { 0, 0, 0, 0 };
#if __clang_major__ >= 23
  v8i z8 = { 0, 0, 0, 0, 0, 0, 0, 0 };
  __builtin_amdgcn_tensor_load_to_lds(g0, g1, g2, g3, z8, 0);
#else
  __builtin_amdgcn_tensor_load_to_lds(g0, g1, g2, g3, 0);
#endif
}
#else
// Fallback: cooperative staged copy (coalesced 16B chunks) with pitch pad.
static __device__ __forceinline__ void stage_full(bf16_t* TtL, const bf16_t* src,
                                                  size_t pstride, int tid) {
  for (int c = tid; c < 2 * D * 32; c += 256) {
    int jj = c & 31;
    int n  = (c >> 5) & (D - 1);
    int p  = c >> 13;
    const v8bf* s = reinterpret_cast<const v8bf*>(src + (size_t)p * pstride +
                                                  (size_t)n * D + jj * 8);
    *reinterpret_cast<v8bf*>(TtL + (p * D + n) * TPITCH + jj * 8) = *s;
  }
}
#endif

// One chain step: acc_p = carry(16x256) @ Tp(256x256), p=0,1; then
// dst[M][n] = x0[M]*acc0 + x1[M]*acc1 (bf16). dst may be LDS (next carry) or global.
static __device__ __forceinline__ void chain_step(bf16_t* TtL, const bf16_t* carryCur,
                                                  const float* xls, const bf16_t* src,
                                                  size_t pstride, bf16_t* dst, size_t dpitch,
                                                  int tid, int lane, int wave) {
#ifdef TDM_OK
  if (wave == 0) {                      // one DMA per workgroup; EXEC-independent issue
    tdm_load_T(src, pstride, 0u);       // TtL sits at dynamic-LDS base (offset 0)
    __builtin_amdgcn_s_wait_tensorcnt(0);
  }
#else
  stage_full(TtL, src, pstride, tid);
#endif
  __syncthreads();

  v8f acc0[2], acc1[2];
#pragma unroll
  for (int i = 0; i < 2; ++i)
#pragma unroll
    for (int e = 0; e < 8; ++e) { acc0[i][e] = 0.f; acc1[i][e] = 0.f; }

  const int Ma  = lane & 15;            // A-matrix row for this lane
  const int kbA = (lane >> 4) << 4;     // A: lanes 16-31 hold K=16..31
  const int kbB = (lane >> 4) << 4;     // B: lanes 16-31 hold K=16..31

#pragma unroll
  for (int i = 0; i < 2; ++i) {
    const int nt = wave * 2 + i;        // this wave's output N-tile
    const bf16_t* arow  = carryCur + Ma * CPITCH;
    const bf16_t* b0row = TtL + (nt * 16 + (lane & 15)) * TPITCH;
    const bf16_t* b1row = b0row + D * TPITCH;
#pragma unroll
    for (int kc = 0; kc < 8; ++kc) {    // 8 x K=32 over full bond dim
      const int k0 = kc * 32;
      v16bf a  = frag2(arow + k0 + kbA,  arow + k0 + kbA + 8);
      v16bf b0 = frag2(b0row + k0 + kbB, b0row + k0 + kbB + 8);
      v16bf b1 = frag2(b1row + k0 + kbB, b1row + k0 + kbB + 8);
      acc0[i] = __builtin_amdgcn_wmma_f32_16x16x32_bf16(false, a, false, b0,
                                                        (short)0, acc0[i], false, false);
      acc1[i] = __builtin_amdgcn_wmma_f32_16x16x32_bf16(false, a, false, b1,
                                                        (short)0, acc1[i], false, false);
    }
  }
  // C/D layout: VGPR r holds row M = r + 8*(lane>=16), column N = lane&15.
  const int hh = lane >> 4, nc = lane & 15;
#pragma unroll
  for (int i = 0; i < 2; ++i) {
    const int nt = wave * 2 + i;
#pragma unroll
    for (int r = 0; r < 8; ++r) {
      const int M = r + 8 * hh;
      float dv = xls[M] * acc0[i][r] + xls[16 + M] * acc1[i][r];
      dst[(size_t)M * dpitch + nt * 16 + nc] = (bf16_t)dv;
    }
  }
}

// --- one-time weight convert/transpose kernels ------------------------------
__global__ void cvt_tmid_kernel(const float* __restrict__ t_mid, bf16_t* __restrict__ Wt) {
  const int total = NMID * 2 * D * D;   // Wt[(s*2+p)][n][j] = t_mid[s][p][j][n]
  for (int o = blockIdx.x * blockDim.x + threadIdx.x; o < total; o += gridDim.x * blockDim.x) {
    int j = o & 255, n = (o >> 8) & 255, sp = o >> 16;
    Wt[o] = (bf16_t)t_mid[((size_t)sp * D + j) * D + n];
  }
}

__global__ void cvt_tlabel_kernel(const float* __restrict__ t_label, bf16_t* __restrict__ Lt) {
  const int total = 2 * NLAB * D * D;   // Lt[(p*10+l)][n=k_out][j] = t_label[p][j][k_out][l]
  for (int o = blockIdx.x * blockDim.x + threadIdx.x; o < total; o += gridDim.x * blockDim.x) {
    int j = o & 255, n = (o >> 8) & 255, pl = o >> 16;
    int l = pl % NLAB, p = pl / NLAB;
    Lt[o] = (bf16_t)t_label[(((size_t)p * D + j) * D + n) * NLAB + l];
  }
}

// --- pre-label chain (sites 0..127) + label fan-out -------------------------
__global__ void __launch_bounds__(256)
mps_chain_pre(const float* __restrict__ inputs, const float* __restrict__ t_first,
              const bf16_t* __restrict__ Wt, const bf16_t* __restrict__ Lt,
              bf16_t* __restrict__ C2) {
  extern __shared__ char smem_raw[];
  bf16_t* TtL    = (bf16_t*)smem_raw;
  bf16_t* carryL = TtL + 2 * D * TPITCH;
  float*  xls    = (float*)(carryL + 2 * 16 * CPITCH);
  const int tid = threadIdx.x, lane = tid & 31, wave = tid >> 5, slab = blockIdx.x;

  { // site 0: left = inputs[:,0] @ t_first
    int b = tid >> 4, ks = (tid & 15) * 16;
    size_t bg = (size_t)slab * 16 + b;
    float x0 = inputs[(bg * D) * 2 + 0];
    float x1 = inputs[(bg * D) * 2 + 1];
    for (int k = ks; k < ks + 16; ++k)
      carryL[b * CPITCH + k] = (bf16_t)(x0 * t_first[k] + x1 * t_first[D + k]);
  }
  __syncthreads();

  int cur = 0;
  for (int s = 0; s < NSTEP_PRE; ++s) {
    const int site = s + 1;
    if (tid < 32) {
      int b = tid & 15, p = tid >> 4;
      xls[p * 16 + b] = inputs[(((size_t)slab * 16 + b) * D + site) * 2 + p];
    }
    const bf16_t* src = Wt + (size_t)s * (2 * D * D);
    if (s + 1 < NSTEP_PRE)                 // pull next step's cores toward L2
      __builtin_prefetch(src + 2 * D * D + tid * 512, 0, 1);
    chain_step(TtL, carryL + cur * 16 * CPITCH, xls, src, (size_t)D * D,
               carryL + (cur ^ 1) * 16 * CPITCH, CPITCH, tid, lane, wave);
    __syncthreads();
    cur ^= 1;
  }

  // label site (128): fan carry out over 10 label GEMMs, write to workspace
  if (tid < 32) {
    int b = tid & 15, p = tid >> 4;
    xls[p * 16 + b] = inputs[(((size_t)slab * 16 + b) * D + 128) * 2 + p];
  }
  __syncthreads();
  for (int l = 0; l < NLAB; ++l) {
    const bf16_t* src = Lt + (size_t)l * (D * D);
    bf16_t* dst = C2 + ((size_t)l * 1024 + (size_t)slab * 16) * D;
    chain_step(TtL, carryL + cur * 16 * CPITCH, xls, src, (size_t)NLAB * D * D,
               dst, (size_t)D, tid, lane, wave);
    __syncthreads();
  }
}

// --- post-label chain (sites 129..254) + final contraction ------------------
__global__ void __launch_bounds__(256)
mps_chain_post(const float* __restrict__ inputs, const float* __restrict__ t_last,
               const bf16_t* __restrict__ Wt, const bf16_t* __restrict__ C2,
               float* __restrict__ out) {
  extern __shared__ char smem_raw[];
  bf16_t* TtL    = (bf16_t*)smem_raw;
  bf16_t* carryL = TtL + 2 * D * TPITCH;
  float*  xls    = (float*)(carryL + 2 * 16 * CPITCH);
  float*  red    = xls + 32;
  const int tid = threadIdx.x, lane = tid & 31, wave = tid >> 5;
  const int l = blockIdx.x >> 6, slab = blockIdx.x & 63;

  { // load label-fanned carry slab
    int b = tid >> 4, ks = (tid & 15) * 16;
    const bf16_t* srcc = C2 + ((size_t)l * 1024 + (size_t)slab * 16 + b) * D;
    for (int k = ks; k < ks + 16; ++k)
      carryL[b * CPITCH + k] = srcc[k];
  }
  __syncthreads();

  int cur = 0;
  for (int j = 0; j < NSTEP_POST; ++j) {
    const int s = NSTEP_PRE + j;          // t_mid[127..252]
    const int site = 129 + j;             // inputs sites 129..254
    if (tid < 32) {
      int b = tid & 15, p = tid >> 4;
      xls[p * 16 + b] = inputs[(((size_t)slab * 16 + b) * D + site) * 2 + p];
    }
    const bf16_t* src = Wt + (size_t)s * (2 * D * D);
    if (j + 1 < NSTEP_POST)
      __builtin_prefetch(src + 2 * D * D + tid * 512, 0, 1);
    chain_step(TtL, carryL + cur * 16 * CPITCH, xls, src, (size_t)D * D,
               carryL + (cur ^ 1) * 16 * CPITCH, CPITCH, tid, lane, wave);
    __syncthreads();
    cur ^= 1;
  }

  // last site (255): out[b,l] = sum_k carry[b,k] * (x0*t_last[0,k] + x1*t_last[1,k])
  const int b = tid >> 4, kk = tid & 15;
  const size_t bg = (size_t)slab * 16 + b;
  const float x0 = inputs[(bg * D + 255) * 2 + 0];
  const float x1 = inputs[(bg * D + 255) * 2 + 1];
  const bf16_t* crow = carryL + cur * 16 * CPITCH + b * CPITCH;
  float part = 0.f;
  for (int k = kk * 16; k < kk * 16 + 16; ++k)
    part += (float)crow[k] * (x0 * t_last[k] + x1 * t_last[D + k]);
  red[b * 16 + kk] = part;
  __syncthreads();
  if (kk == 0) {
    float sum = 0.f;
    for (int i2 = 0; i2 < 16; ++i2) sum += red[b * 16 + i2];
    out[bg * NLAB + l] = sum;
  }
}

// ---------------------------------------------------------------------------
extern "C" void kernel_launch(void* const* d_in, const int* in_sizes, int n_in,
                              void* d_out, int out_size, void* d_ws, size_t ws_size,
                              hipStream_t stream) {
  (void)in_sizes; (void)n_in; (void)out_size; (void)ws_size;
  const float* inputs  = (const float*)d_in[0];   // (1024, 256, 2)
  const float* t_first = (const float*)d_in[1];   // (2, 256)
  const float* t_mid   = (const float*)d_in[2];   // (253, 2, 256, 256)
  const float* t_label = (const float*)d_in[3];   // (2, 256, 256, 10)
  const float* t_last  = (const float*)d_in[4];   // (2, 256)
  float* out = (float*)d_out;                     // (1024, 10) f32

  // Workspace: bf16 transposed cores (66.3MB) | bf16 label cores (2.6MB) | label-fanned carry (5.2MB)
  bf16_t* Wt = (bf16_t*)d_ws;
  bf16_t* Lt = Wt + (size_t)NMID * 2 * D * D;
  bf16_t* C2 = Lt + (size_t)2 * NLAB * D * D;

  cvt_tmid_kernel  <<<8192, 256, 0, stream>>>(t_mid, Wt);
  cvt_tlabel_kernel<<<1280, 256, 0, stream>>>(t_label, Lt);
  mps_chain_pre    <<<64,   256, SMEM_BYTES, stream>>>(inputs, t_first, Wt, Lt, C2);
  mps_chain_post   <<<640,  256, SMEM_BYTES, stream>>>(inputs, t_last, Wt, C2, out);
}
